// GAT_19610820673943
// MI455X (gfx1250) — matrix operand order repaired and checked by reading, not compiled, not run on hardware
//
#include <hip/hip_runtime.h>

// ---------------------------------------------------------------------------
// GAT (2-layer) pipeline for MI455X (gfx1250, wave32).
// GEMMs run on v_wmma_f32_16x16x32_f16 (f16 in, f32 accumulate).
// Edge softmax/aggregation uses native f32/u32 global atomics (relaxed,agent).
// ---------------------------------------------------------------------------

typedef __attribute__((ext_vector_type(16))) _Float16 v16h;
typedef __attribute__((ext_vector_type(8)))  _Float16 v8h;
typedef __attribute__((ext_vector_type(8)))  float    v8f;

#define N_NODES 50000
#define N_EDGES 800000
#define EPLUS   (N_EDGES + N_NODES)   // edges + self loops
#define NGRAPH  64
#define IN_DIM  128
#define HID     64
#define HEADS   4
#define C1      (HEADS * HID)         // 256
#define OUT_DIM 32
#define NEG_SLOPE 0.2f

// ---- helpers --------------------------------------------------------------

static __device__ __forceinline__ unsigned f32_key(float v) {
  unsigned b = __float_as_uint(v);
  return (b & 0x80000000u) ? ~b : (b | 0x80000000u);   // order-preserving map
}
static __device__ __forceinline__ float f32_unkey(unsigned k) {
  unsigned b = (k & 0x80000000u) ? (k & 0x7fffffffu) : ~k;
  return __uint_as_float(b);
}
static __device__ __forceinline__ void atomAddF(float* p, float v) {
  __hip_atomic_fetch_add(p, v, __ATOMIC_RELAXED, __HIP_MEMORY_SCOPE_AGENT);
}
static __device__ __forceinline__ void atomMaxU(unsigned* p, unsigned v) {
  __hip_atomic_fetch_max(p, v, __ATOMIC_RELAXED, __HIP_MEMORY_SCOPE_AGENT);
}

// ---- init / convert kernels ----------------------------------------------

__global__ void k_fill_u32(unsigned* p, unsigned val, int n) {
  int i = blockIdx.x * blockDim.x + threadIdx.x;
  if (i < n) p[i] = val;
}

__global__ void k_f32_to_f16(const float* __restrict__ in,
                             _Float16* __restrict__ out, int n) {
  int i = blockIdx.x * blockDim.x + threadIdx.x;
  if (i < n) out[i] = (_Float16)in[i];
}

// Pack W [K x Ncols] (row-major f32) into per-lane WMMA B-fragments:
// Bp[((tn*Ksteps + k0)*32 + lane)*16 + i] = W[(k0*32 + (lane>>4)*16 + i)*Ncols
//                                             + tn*16 + (lane&15)]
__global__ void k_pack_B(const float* __restrict__ W, _Float16* __restrict__ Bp,
                         int Ksteps, int Ntiles, int Ncols) {
  int idx = blockIdx.x * blockDim.x + threadIdx.x;
  int total = Ntiles * Ksteps * 32 * 16;
  if (idx >= total) return;
  int i    = idx & 15;
  int lane = (idx >> 4) & 31;
  int rem  = idx >> 9;
  int k0   = rem % Ksteps;
  int tn   = rem / Ksteps;
  int Krow = k0 * 32 + ((lane >> 4) << 4) + i;
  int ncol = tn * 16 + (lane & 15);
  Bp[idx] = (_Float16)W[(size_t)Krow * Ncols + ncol];
}

// ---- WMMA GEMM: C[M x Ncols] = A[M x Kdim](f16,row-major) @ Bpack ---------
// one wave per 16x16 output tile; M assumed multiple of 16 (50000 = 3125*16)

__global__ void k_wmma_gemm(const _Float16* __restrict__ A,
                            const _Float16* __restrict__ Bp,
                            float* __restrict__ C,
                            int Mtiles, int Ntiles, int Ksteps,
                            int Kdim, int Ncols) {
  int wave = (blockIdx.x * blockDim.x + threadIdx.x) >> 5;
  int lane = threadIdx.x & 31;
  if (wave >= Mtiles * Ntiles) return;              // wave-uniform: EXEC stays full
  int tm = wave / Ntiles;
  int tn = wave - tm * Ntiles;

  int arow  = tm * 16 + (lane & 15);
  int kbase = (lane >> 4) * 16;                     // contiguous K per lane half (matches Bpack)
  const _Float16* ap = A + (size_t)arow * Kdim;
  const _Float16* bp = Bp + ((size_t)tn * Ksteps * 32 + lane) * 16;

  v8f c = {};
  for (int k0 = 0; k0 < Ksteps; ++k0) {
    if (k0 + 1 < Ksteps)                            // -> global_prefetch_b8
      __builtin_prefetch(ap + (k0 + 1) * 32 + kbase, 0, 1);
    v8h lo = *(const v8h*)(ap + k0 * 32 + kbase);        // K = kbase .. kbase+7
    v8h hi = *(const v8h*)(ap + k0 * 32 + kbase + 8);    // K = kbase+8 .. +15
    v16h a = __builtin_shufflevector(lo, hi, 0,1,2,3,4,5,6,7,
                                             8,9,10,11,12,13,14,15);
    v16h b = *(const v16h*)(bp + (size_t)k0 * 32 * 16);
    c = __builtin_amdgcn_wmma_f32_16x16x32_f16(false, a, false, b,
                                               (short)0, c, false, false);
  }
  // C/D layout: lane 0-15 -> N=lane, M=r; lane 16-31 -> N=lane-16, M=8+r
  int col   = tn * 16 + (lane & 15);
  int rbase = tm * 16 + (lane >> 4) * 8;
  for (int r = 0; r < 8; ++r)
    C[(size_t)(rbase + r) * Ncols + col] = c[r];
}

// ---- per-node attention coefficients: as/ad [N*H] -------------------------

__global__ void k_node_alpha(const float* __restrict__ h,
                             const float* __restrict__ a_src,
                             const float* __restrict__ a_dst,
                             float* __restrict__ as, float* __restrict__ ad,
                             int Nn, int H, int D) {
  int idx = blockIdx.x * blockDim.x + threadIdx.x;
  if (idx >= Nn * H) return;
  int n  = idx / H;
  int hh = idx - n * H;
  const float* hp = h + ((size_t)n * H + hh) * D;
  float s = 0.f, d = 0.f;
  for (int k = 0; k < D; ++k) {
    float v = hp[k];
    s += v * a_src[hh * D + k];
    d += v * a_dst[hh * D + k];
  }
  as[idx] = s;
  ad[idx] = d;
}

// ---- edge pass 1: leaky-relu score + segment max (ordered-u32 atomic) -----

__global__ void k_edge_max(const long long* __restrict__ adj,
                           const float* __restrict__ as,
                           const float* __restrict__ ad,
                           float* __restrict__ eraw, unsigned* __restrict__ mmax,
                           int H) {
  int idx = blockIdx.x * blockDim.x + threadIdx.x;
  if (idx >= EPLUS * H) return;
  int e = idx / H;
  int h = idx - e * H;
  int s, d;
  if (e < N_EDGES) { s = (int)adj[e]; d = (int)adj[N_EDGES + e]; }
  else             { s = d = e - N_EDGES; }
  float v = as[(size_t)s * H + h] + ad[(size_t)d * H + h];
  v = (v >= 0.f) ? v : NEG_SLOPE * v;
  eraw[idx] = v;
  atomMaxU(&mmax[(size_t)d * H + h], f32_key(v));
}

// ---- edge pass 2: exp(e - m[dst]) in place + segment denom ---------------

__global__ void k_edge_expsum(const long long* __restrict__ adj,
                              const unsigned* __restrict__ mmax,
                              float* __restrict__ ex,   // in: eraw, out: exp
                              float* __restrict__ den, int H) {
  int idx = blockIdx.x * blockDim.x + threadIdx.x;
  if (idx >= EPLUS * H) return;
  int e = idx / H;
  int h = idx - e * H;
  int d = (e < N_EDGES) ? (int)adj[N_EDGES + e] : (e - N_EDGES);
  float m = f32_unkey(mmax[(size_t)d * H + h]);
  float x = __expf(ex[idx] - m);
  ex[idx] = x;
  atomAddF(&den[(size_t)d * H + h], x);
}

// ---- edge pass 3: agg[dst, c] += feat[src, c] * alpha (wave per edge) -----

__global__ void k_edge_agg(const long long* __restrict__ adj,
                           const float* __restrict__ feat,
                           const float* __restrict__ ex,
                           const float* __restrict__ den,
                           float* __restrict__ agg, int H, int C) {
  int gid  = blockIdx.x * blockDim.x + threadIdx.x;
  int wave = gid >> 5;
  int lane = gid & 31;
  if (wave >= EPLUS) return;                        // wave-uniform
  int s, d;
  if (wave < N_EDGES) { s = (int)adj[wave]; d = (int)adj[N_EDGES + wave]; }
  else                { s = d = wave - N_EDGES; }
  int D = C / H;
  for (int c = lane; c < C; c += 32) {
    int h = c / D;
    float alpha = ex[(size_t)wave * H + h] / den[(size_t)d * H + h];
    float v = feat[(size_t)s * C + c] * alpha;
    atomAddF(&agg[(size_t)d * C + c], v);
  }
}

// ---- bias + relu + f16 convert -------------------------------------------

__global__ void k_bias_relu_f16(const float* __restrict__ agg,
                                const float* __restrict__ bias,
                                _Float16* __restrict__ out, int total, int C) {
  int i = blockIdx.x * blockDim.x + threadIdx.x;
  if (i >= total) return;
  float v = agg[i] + bias[i % C];
  out[i] = (_Float16)fmaxf(v, 0.f);
}

// ---- final per-node: +b2, log_softmax(32), pooled segment sums ------------
// wave per node: lane = channel (0..31), wave32 shuffle reductions

__global__ void k_node_lsm_pool(const float* __restrict__ agg2,
                                const float* __restrict__ b2,
                                const long long* __restrict__ batch,
                                float* __restrict__ pooled,
                                float* __restrict__ counts) {
  int gid  = blockIdx.x * blockDim.x + threadIdx.x;
  int wave = gid >> 5;
  int lane = gid & 31;
  if (wave >= N_NODES) return;
  float v = agg2[(size_t)wave * OUT_DIM + lane] + b2[lane];
  float m = v;
  for (int o = 16; o > 0; o >>= 1) m = fmaxf(m, __shfl_xor(m, o, 32));
  float e = __expf(v - m);
  float ssum = e;
  for (int o = 16; o > 0; o >>= 1) ssum += __shfl_xor(ssum, o, 32);
  float lsm = v - m - __logf(ssum);
  int g = (int)batch[wave];
  atomAddF(&pooled[(size_t)g * OUT_DIM + lane], lsm);
  if (lane == 0) atomAddF(&counts[g], 1.0f);
}

__global__ void k_final_linear(const float* __restrict__ pooled,
                               const float* __restrict__ counts,
                               const float* __restrict__ linW,
                               const float* __restrict__ linb,
                               float* __restrict__ out) {
  int g = blockIdx.x * blockDim.x + threadIdx.x;
  if (g >= NGRAPH) return;
  float inv = 1.0f / fmaxf(counts[g], 1.0f);
  float acc = 0.f;
  for (int c = 0; c < OUT_DIM; ++c)
    acc += pooled[(size_t)g * OUT_DIM + c] * inv * linW[c];
  out[g] = acc + linb[0];
}

// ---------------------------------------------------------------------------

extern "C" void kernel_launch(void* const* d_in, const int* in_sizes, int n_in,
                              void* d_out, int out_size, void* d_ws, size_t ws_size,
                              hipStream_t stream) {
  (void)in_sizes; (void)n_in; (void)out_size; (void)ws_size;

  const float*     x      = (const float*)d_in[0];
  const long long* adj    = (const long long*)d_in[1];
  const long long* batch  = (const long long*)d_in[2];
  const float*     W1     = (const float*)d_in[3];
  const float*     a1_src = (const float*)d_in[4];
  const float*     a1_dst = (const float*)d_in[5];
  const float*     b1     = (const float*)d_in[6];
  const float*     W2     = (const float*)d_in[7];
  const float*     a2_src = (const float*)d_in[8];
  const float*     a2_dst = (const float*)d_in[9];
  const float*     b2     = (const float*)d_in[10];
  const float*     lin_W  = (const float*)d_in[11];
  const float*     lin_b  = (const float*)d_in[12];
  float* out = (float*)d_out;

  // -------- workspace layout (bytes, 256-aligned) --------
  char* ws = (char*)d_ws;
  size_t off = 0;
  auto alloc = [&](size_t bytes) -> char* {
    char* p = ws + off;
    off += (bytes + 255) & ~(size_t)255;
    return p;
  };
  _Float16* xf16  = (_Float16*)alloc((size_t)N_NODES * IN_DIM * 2);
  _Float16* w1p   = (_Float16*)alloc(16 * 4 * 32 * 16 * 2);
  _Float16* w2p   = (_Float16*)alloc(2 * 8 * 32 * 16 * 2);
  float*    h1    = (float*)   alloc((size_t)N_NODES * C1 * 4);
  float*    as1   = (float*)   alloc((size_t)N_NODES * HEADS * 4);
  float*    ad1   = (float*)   alloc((size_t)N_NODES * HEADS * 4);
  float*    ex1   = (float*)   alloc((size_t)EPLUS * HEADS * 4);  // eraw -> exp
  unsigned* m1    = (unsigned*)alloc((size_t)N_NODES * HEADS * 4);
  float*    den1  = (float*)   alloc((size_t)N_NODES * HEADS * 4);
  float*    agg1  = (float*)   alloc((size_t)N_NODES * C1 * 4);
  _Float16* h1f16 = (_Float16*)alloc((size_t)N_NODES * C1 * 2);
  float*    h2    = (float*)   alloc((size_t)N_NODES * OUT_DIM * 4);
  float*    as2   = (float*)   alloc((size_t)N_NODES * 4);
  float*    ad2   = (float*)   alloc((size_t)N_NODES * 4);
  float*    ex2   = (float*)   alloc((size_t)EPLUS * 4);
  unsigned* m2    = (unsigned*)alloc((size_t)N_NODES * 4);
  float*    den2  = (float*)   alloc((size_t)N_NODES * 4);
  float*    agg2  = (float*)   alloc((size_t)N_NODES * OUT_DIM * 4);
  float*    pooled= (float*)   alloc((size_t)NGRAPH * OUT_DIM * 4);
  float*    counts= (float*)   alloc((size_t)NGRAPH * 4);

  auto blocks = [](long long n, int tpb) { return (unsigned)((n + tpb - 1) / tpb); };
  const unsigned NEG_INF_KEY = 0x007FFFFFu;  // f32_key(-inf)

  // -------- init --------
  hipMemsetAsync(den1,   0, (size_t)N_NODES * HEADS * 4, stream);
  hipMemsetAsync(agg1,   0, (size_t)N_NODES * C1 * 4, stream);
  hipMemsetAsync(den2,   0, (size_t)N_NODES * 4, stream);
  hipMemsetAsync(agg2,   0, (size_t)N_NODES * OUT_DIM * 4, stream);
  hipMemsetAsync(pooled, 0, (size_t)NGRAPH * OUT_DIM * 4, stream);
  hipMemsetAsync(counts, 0, (size_t)NGRAPH * 4, stream);
  k_fill_u32<<<blocks(N_NODES * HEADS, 256), 256, 0, stream>>>(m1, NEG_INF_KEY, N_NODES * HEADS);
  k_fill_u32<<<blocks(N_NODES, 256), 256, 0, stream>>>(m2, NEG_INF_KEY, N_NODES);

  // -------- f16 conversions + B packs --------
  k_f32_to_f16<<<blocks((long long)N_NODES * IN_DIM, 256), 256, 0, stream>>>(
      x, xf16, N_NODES * IN_DIM);
  k_pack_B<<<blocks(16 * 4 * 32 * 16, 256), 256, 0, stream>>>(W1, w1p, 4, 16, C1);
  k_pack_B<<<blocks(2 * 8 * 32 * 16, 256), 256, 0, stream>>>(W2, w2p, 8, 2, OUT_DIM);

  // -------- layer 1: GEMM1 (3125x16 tiles, K=128) --------
  {
    long long waves = 3125LL * 16;
    k_wmma_gemm<<<blocks(waves * 32, 256), 256, 0, stream>>>(
        xf16, w1p, h1, 3125, 16, 4, IN_DIM, C1);
  }
  k_node_alpha<<<blocks((long long)N_NODES * HEADS, 256), 256, 0, stream>>>(
      h1, a1_src, a1_dst, as1, ad1, N_NODES, HEADS, HID);
  k_edge_max<<<blocks((long long)EPLUS * HEADS, 256), 256, 0, stream>>>(
      adj, as1, ad1, ex1, m1, HEADS);
  k_edge_expsum<<<blocks((long long)EPLUS * HEADS, 256), 256, 0, stream>>>(
      adj, m1, ex1, den1, HEADS);
  k_edge_agg<<<blocks((long long)EPLUS * 32, 256), 256, 0, stream>>>(
      adj, h1, ex1, den1, agg1, HEADS, C1);
  k_bias_relu_f16<<<blocks((long long)N_NODES * C1, 256), 256, 0, stream>>>(
      agg1, b1, h1f16, N_NODES * C1, C1);

  // -------- layer 2: GEMM2 (3125x2 tiles, K=256) --------
  {
    long long waves = 3125LL * 2;
    k_wmma_gemm<<<blocks(waves * 32, 256), 256, 0, stream>>>(
        h1f16, w2p, h2, 3125, 2, 8, C1, OUT_DIM);
  }
  k_node_alpha<<<blocks((long long)N_NODES, 256), 256, 0, stream>>>(
      h2, a2_src, a2_dst, as2, ad2, N_NODES, 1, OUT_DIM);
  k_edge_max<<<blocks((long long)EPLUS, 256), 256, 0, stream>>>(
      adj, as2, ad2, ex2, m2, 1);
  k_edge_expsum<<<blocks((long long)EPLUS, 256), 256, 0, stream>>>(
      adj, m2, ex2, den2, 1);
  k_edge_agg<<<blocks((long long)EPLUS * 32, 256), 256, 0, stream>>>(
      adj, h2, ex2, den2, agg2, 1, OUT_DIM);

  // -------- log_softmax + mean pool + final linear --------
  k_node_lsm_pool<<<blocks((long long)N_NODES * 32, 256), 256, 0, stream>>>(
      agg2, b2, batch, pooled, counts);
  k_final_linear<<<1, 64, 0, stream>>>(pooled, counts, lin_W, lin_b, out);
}
